// ViGBlock_67242007986279
// MI455X (gfx1250) — matrix operand order repaired and checked by reading, not compiled, not run on hardware
//
#include <hip/hip_runtime.h>
#include <hip/hip_bf16.h>

typedef __attribute__((ext_vector_type(16))) _Float16 v16h;
typedef __attribute__((ext_vector_type(8)))  _Float16 v8h;
typedef __attribute__((ext_vector_type(8)))  float    v8f;

#define BB   8
#define CC   96
#define NPIX 3136      // 56*56 = 196 * 16
#define KNN  9
#define EPSI 1e-5f

// ---------------------------------------------------------------------------
// fp32 -> f16 weight conversion (weights stay (Cout, Cin) row-major: that is
// already the contiguous-per-lane layout for the B fragment).
// ---------------------------------------------------------------------------
__global__ void cvt_f16_kernel(const float* __restrict__ src, _Float16* __restrict__ dst, int n) {
    int i = blockIdx.x * blockDim.x + threadIdx.x;
    if (i < n) dst[i] = (_Float16)src[i];
}

// ---------------------------------------------------------------------------
// Transpose + convert activations: X (B,Cc,N) fp32 -> XT (B,N,Cc) f16.
// Reads coalesced along N; each thread writes Cc contiguous halves.
// This makes every WMMA fragment load a contiguous b128 per lane.
// ---------------------------------------------------------------------------
__global__ void transpose_f16_kernel(const float* __restrict__ X, _Float16* __restrict__ XT,
                                     int Cc) {
    int t = blockIdx.x * blockDim.x + threadIdx.x;   // b*NPIX + n
    if (t >= BB * NPIX) return;
    int b = t / NPIX, n = t % NPIX;
    const float* p = X + (size_t)b * Cc * NPIX + n;
    _Float16* q = XT + (size_t)t * Cc;
    for (int c = 0; c < Cc; ++c) q[c] = (_Float16)p[(size_t)c * NPIX];
}

// ---------------------------------------------------------------------------
// 1x1 conv as GEMM:  Out[b, oc, n] = sum_c W[oc, c] * XT[b, n, c] + bias[oc]
// One wave32 per 16(pixel) x 16(out-ch) tile, K-steps of 32 via WMMA.
// A fragment (16-bit 16x32, k order chosen here, same as B): lane L: M=L&15, elem e -> K=(L>=16?16:0)+e
//   -> from row-major XT: one contiguous v16h (2x b128) load.
// B fragment (16-bit 32x16, k order chosen here, same as A): lane L: N=L&15, elem e -> K=(L>=16?16:0)+e
//   -> from row-major W16: one contiguous v16h (2x b128) load.
// C/D: elem v -> M = v + (L>=16?8:0), N = L&15.
// ---------------------------------------------------------------------------
__global__ __launch_bounds__(32)
void conv1x1_wmma_kernel(const _Float16* __restrict__ XT, const _Float16* __restrict__ W16,
                         const float* __restrict__ bias, float* __restrict__ Out,
                         int Cin, int Cout) {
    const int lane = threadIdx.x;
    const int b    = blockIdx.z;
    const int row0 = blockIdx.x * 16;   // pixel tile
    const int col0 = blockIdx.y * 16;   // out-channel tile
    const int m    = lane & 15;
    const bool hi  = lane >= 16;

    const _Float16* Ab = XT + ((size_t)b * NPIX + row0 + m) * Cin;
    const _Float16* Wb = W16 + (size_t)(col0 + m) * Cin;

    v8f acc = {};
    for (int kk = 0; kk < Cin; kk += 32) {
        v16h a = *(const v16h*)(Ab + kk + (hi ? 16 : 0));
        v16h bm = *(const v16h*)(Wb + kk + (hi ? 16 : 0));
        acc = __builtin_amdgcn_wmma_f32_16x16x32_f16(false, a, false, bm,
                                                     (short)0, acc, false, false);
    }

    const int oc = col0 + m;
    const float bv = bias ? bias[oc] : 0.0f;
    float* outp = Out + ((size_t)b * Cout + oc) * NPIX + row0 + (hi ? 8 : 0);
#pragma unroll
    for (int v = 0; v < 8; ++v) outp[v] = acc[v] + bv;
}

// ---------------------------------------------------------------------------
// InstanceNorm over spatial dim per (b, channel), fused activation and
// optional residual:  out = act((x-mean)*rsqrt(var+eps)) + res
// act: 0=none, 1=gelu(exact), 2=relu.  One 256-thread block per (b,channel).
// ---------------------------------------------------------------------------
__global__ __launch_bounds__(256)
void inorm_kernel(const float* __restrict__ In, float* __restrict__ Out,
                  const float* __restrict__ Res, int act) {
    const int bc = blockIdx.x;
    const float* p = In + (size_t)bc * NPIX;
    float s = 0.0f, s2 = 0.0f;
    for (int i = threadIdx.x; i < NPIX; i += 256) {
        float v = p[i];
        s += v; s2 += v * v;
    }
    __shared__ float sh[256], sh2[256];
    sh[threadIdx.x] = s; sh2[threadIdx.x] = s2;
    __syncthreads();
    for (int st = 128; st > 0; st >>= 1) {
        if (threadIdx.x < st) {
            sh[threadIdx.x]  += sh[threadIdx.x + st];
            sh2[threadIdx.x] += sh2[threadIdx.x + st];
        }
        __syncthreads();
    }
    const float mean = sh[0] * (1.0f / NPIX);
    const float var  = sh2[0] * (1.0f / NPIX) - mean * mean;
    const float inv  = rsqrtf(var + EPSI);

    float* q = Out + (size_t)bc * NPIX;
    const float* r = Res ? (Res + (size_t)bc * NPIX) : nullptr;
    for (int i = threadIdx.x; i < NPIX; i += 256) {
        float v = (p[i] - mean) * inv;
        if (act == 1)      v = 0.5f * v * (1.0f + erff(v * 0.70710678118654752f));
        else if (act == 2) v = fmaxf(v, 0.0f);
        if (r) v += r[i];
        q[i] = v;
    }
}

// ---------------------------------------------------------------------------
// Per-column l2 normalization, emitted TRANSPOSED:
//   XnT[b, n, c] = f16( X[b,c,n] / max(||X[b,:,n]||, 1e-12) )  (B,N,C) f16
//   Sq[b, n]     = sum_c Xn^2 (fp32)
// ---------------------------------------------------------------------------
__global__ void colnorm_kernel(const float* __restrict__ Xf, _Float16* __restrict__ XnT,
                               float* __restrict__ Sq) {
    int idx = blockIdx.x * blockDim.x + threadIdx.x;   // b*NPIX + n
    if (idx >= BB * NPIX) return;
    int b = idx / NPIX, n = idx % NPIX;
    const float* p = Xf + (size_t)b * CC * NPIX + n;
    float s2 = 0.0f;
    for (int c = 0; c < CC; ++c) { float v = p[(size_t)c * NPIX]; s2 += v * v; }
    float inv = 1.0f / fmaxf(sqrtf(s2), 1e-12f);
    _Float16* q = XnT + (size_t)idx * CC;
    for (int c = 0; c < CC; ++c) q[c] = (_Float16)(p[(size_t)c * NPIX] * inv);
    Sq[idx] = s2 * inv * inv;
}

// ---------------------------------------------------------------------------
// Fused kNN: one wave per 16-row strip. Gram tile via WMMA (C=96 -> 3 k-steps),
// dist = sq_r - 2*S + sq_c + rp, per-row streaming top-9 (smallest).
// A fragments for the strip are register-cached across the whole column sweep;
// B fragments are contiguous v16h loads from XnT; next tile is prefetched.
// The N x N distance matrix is never materialized (saves ~315 MB/grapher).
// ---------------------------------------------------------------------------
__global__ __launch_bounds__(32)
void knn_kernel(const _Float16* __restrict__ XnT, const float* __restrict__ Sq,
                const float* __restrict__ rp, int* __restrict__ Idx) {
    const int lane = threadIdx.x;
    const int b    = blockIdx.y;
    const int row0 = blockIdx.x * 16;
    const int m    = lane & 15;
    const bool hi  = lane >= 16;

    const _Float16* Xb = XnT + (size_t)b * NPIX * CC;
    const float*   sqb = Sq + (size_t)b * NPIX;

    // A fragments (strip rows), cached in registers for the whole sweep.
    v16h A[3];
    {
        const _Float16* Ar = Xb + (size_t)(row0 + m) * CC;
#pragma unroll
        for (int ks = 0; ks < 3; ++ks) {
            A[ks] = *(const v16h*)(Ar + ks * 32 + (hi ? 16 : 0));
        }
    }

    float bd[KNN]; int bi[KNN];
#pragma unroll
    for (int j = 0; j < KNN; ++j) { bd[j] = 3.4e38f; bi[j] = 0; }

    __shared__ float tile[16][17];

    for (int ct = 0; ct < NPIX / 16; ++ct) {
        const int col0 = ct * 16;
        const _Float16* Br = Xb + (size_t)(col0 + m) * CC;
        if (ct + 1 < NPIX / 16)   // gfx1250 global_prefetch_b8 for next B tile
            __builtin_prefetch(Br + 16 * CC, 0, 0);

        v8f acc = {};
#pragma unroll
        for (int ks = 0; ks < 3; ++ks) {
            v16h bm = *(const v16h*)(Br + ks * 32 + (hi ? 16 : 0));
            acc = __builtin_amdgcn_wmma_f32_16x16x32_f16(false, A[ks], false, bm,
                                                         (short)0, acc, false, false);
        }
        const int colg = col0 + m;
        const float sqc = sqb[colg];
#pragma unroll
        for (int v = 0; v < 8; ++v) {
            int rowl = v + (hi ? 8 : 0);
            int rowg = row0 + rowl;
            float d = sqb[rowg] - 2.0f * acc[v] + sqc + rp[(size_t)rowg * NPIX + colg];
            tile[rowl][m] = d;
        }
        __syncthreads();
        if (lane < 16) {
#pragma unroll 1
            for (int c2 = 0; c2 < 16; ++c2) {
                float nd = tile[lane][c2];
                if (nd < bd[KNN - 1]) {          // rare path; strict < keeps low index on ties
                    int ni = col0 + c2;
#pragma unroll
                    for (int j = 0; j < KNN; ++j) {   // register-only bubble insert
                        bool sw = nd < bd[j];
                        float td = sw ? bd[j] : nd;  int ti = sw ? bi[j] : ni;
                        bd[j] = sw ? nd : bd[j];     bi[j] = sw ? ni : bi[j];
                        nd = td; ni = ti;
                    }
                }
            }
        }
        __syncthreads();
    }

    if (lane < 16) {
        int* op = Idx + ((size_t)b * NPIX + row0 + lane) * KNN;
#pragma unroll
        for (int j = 0; j < KNN; ++j) op[j] = bi[j];
    }
}

// ---------------------------------------------------------------------------
// Max-relative graph conv + channel-interleaved concat:
//   Y[b,2c,n] = x_i ; Y[b,2c+1,n] = max_k( x[idx[n,k]] - x_i )
// ---------------------------------------------------------------------------
__global__ void maxrel_kernel(const float* __restrict__ Xf, const int* __restrict__ Idx,
                              float* __restrict__ Y) {
    int t = blockIdx.x * blockDim.x + threadIdx.x;
    if (t >= BB * CC * NPIX) return;
    int n = t % NPIX; int bc = t / NPIX; int c = bc % CC; int b = bc / CC;
    const float* xc = Xf + (size_t)(b * CC + c) * NPIX;
    float xi = xc[n];
    const int* ip = Idx + ((size_t)b * NPIX + n) * KNN;
    float mx = -3.4e38f;
#pragma unroll
    for (int k = 0; k < KNN; ++k) mx = fmaxf(mx, xc[ip[k]] - xi);
    float* yb = Y + (size_t)b * (2 * CC) * NPIX;
    yb[(size_t)(2 * c)     * NPIX + n] = xi;
    yb[(size_t)(2 * c + 1) * NPIX + n] = mx;
}

// ---------------------------------------------------------------------------
// Host-side orchestration
// ---------------------------------------------------------------------------
static void launch_conv(const float* Xin, _Float16* actT, const _Float16* W16,
                        const float* bias, float* Out, int Cin, int Cout,
                        hipStream_t stream) {
    transpose_f16_kernel<<<(BB * NPIX + 255) / 256, 256, 0, stream>>>(Xin, actT, Cin);
    conv1x1_wmma_kernel<<<dim3(NPIX / 16, Cout / 16, BB), 32, 0, stream>>>(actT, W16, bias, Out,
                                                                           Cin, Cout);
}

static void launch_grapher(const float* rp, float* cur, float* bufA, float* bufB,
                           _Float16* actT, _Float16* xnT, float* sq, int* idx,
                           const _Float16* wfc1, const float* bfc1,
                           const _Float16* wmr,  const float* bmr,
                           const _Float16* wfc2, const float* bfc2,
                           hipStream_t stream) {
    // x = inorm(conv(cur, fc1))           -> bufA (B,C,N)
    launch_conv(cur, actT, wfc1, bfc1, bufA, CC, CC, stream);
    inorm_kernel<<<BB * CC, 256, 0, stream>>>(bufA, bufA, nullptr, 0);
    // dynamic kNN on l2-normalized features
    colnorm_kernel<<<(BB * NPIX + 255) / 256, 256, 0, stream>>>(bufA, xnT, sq);
    knn_kernel<<<dim3(NPIX / 16, BB), 32, 0, stream>>>(xnT, sq, rp, idx);
    // max-relative + interleave            -> bufB (B,2C,N)
    maxrel_kernel<<<(BB * CC * NPIX + 255) / 256, 256, 0, stream>>>(bufA, idx, bufB);
    // y = gelu(inorm(conv(y, mr)))        -> bufA (B,2C,N)
    launch_conv(bufB, actT, wmr, bmr, bufA, 2 * CC, 2 * CC, stream);
    inorm_kernel<<<BB * 2 * CC, 256, 0, stream>>>(bufA, bufA, nullptr, 1);
    // y = inorm(conv(y, fc2)) + shortcut  -> cur (B,C,N)
    launch_conv(bufA, actT, wfc2, bfc2, bufB, 2 * CC, CC, stream);
    inorm_kernel<<<BB * CC, 256, 0, stream>>>(bufB, cur, cur, 0);
}

static void launch_ffn(float* cur, float* bufA, float* bufB, _Float16* actT,
                       const _Float16* w1, const float* b1,
                       const _Float16* w2, const float* b2, hipStream_t stream) {
    launch_conv(cur, actT, w1, b1, bufA, CC, 4 * CC, stream);
    inorm_kernel<<<BB * 4 * CC, 256, 0, stream>>>(bufA, bufA, nullptr, 1);
    launch_conv(bufA, actT, w2, b2, bufB, 4 * CC, CC, stream);
    inorm_kernel<<<BB * CC, 256, 0, stream>>>(bufB, cur, cur, 0);
}

extern "C" void kernel_launch(void* const* d_in, const int* in_sizes, int n_in,
                              void* d_out, int out_size, void* d_ws, size_t ws_size,
                              hipStream_t stream) {
    const float* x  = (const float*)d_in[0];
    const float* rp = (const float*)d_in[1];
    const float* g1_fc1_w = (const float*)d_in[2];  const float* g1_fc1_b = (const float*)d_in[3];
    const float* g1_mr_w  = (const float*)d_in[4];  const float* g1_mr_b  = (const float*)d_in[5];
    const float* g1_fc2_w = (const float*)d_in[6];  const float* g1_fc2_b = (const float*)d_in[7];
    const float* g2_fc1_w = (const float*)d_in[8];  const float* g2_fc1_b = (const float*)d_in[9];
    const float* g2_mr_w  = (const float*)d_in[10]; const float* g2_mr_b  = (const float*)d_in[11];
    const float* g2_fc2_w = (const float*)d_in[12]; const float* g2_fc2_b = (const float*)d_in[13];
    const float* f1_w1 = (const float*)d_in[14]; const float* f1_b1 = (const float*)d_in[15];
    const float* f1_w2 = (const float*)d_in[16]; const float* f1_b2 = (const float*)d_in[17];
    const float* f2_w1 = (const float*)d_in[18]; const float* f2_b1 = (const float*)d_in[19];
    const float* f2_w2 = (const float*)d_in[20]; const float* f2_b2 = (const float*)d_in[21];

    // scratch arena
    char* ws = (char*)d_ws;
    size_t off = 0;
    auto alloc = [&](size_t bytes) -> void* {
        void* p = ws + off;
        off = (off + bytes + 255) & ~(size_t)255;
        return p;
    };
    _Float16* w16_g1_fc1 = (_Float16*)alloc((size_t)CC * CC * 2);
    _Float16* w16_g1_mr  = (_Float16*)alloc((size_t)2 * CC * 2 * CC * 2);
    _Float16* w16_g1_fc2 = (_Float16*)alloc((size_t)CC * 2 * CC * 2);
    _Float16* w16_g2_fc1 = (_Float16*)alloc((size_t)CC * CC * 2);
    _Float16* w16_g2_mr  = (_Float16*)alloc((size_t)2 * CC * 2 * CC * 2);
    _Float16* w16_g2_fc2 = (_Float16*)alloc((size_t)CC * 2 * CC * 2);
    _Float16* w16_f1_w1  = (_Float16*)alloc((size_t)4 * CC * CC * 2);
    _Float16* w16_f1_w2  = (_Float16*)alloc((size_t)CC * 4 * CC * 2);
    _Float16* w16_f2_w1  = (_Float16*)alloc((size_t)4 * CC * CC * 2);
    _Float16* w16_f2_w2  = (_Float16*)alloc((size_t)CC * 4 * CC * 2);
    float*    cur  = (float*)alloc((size_t)BB * CC * NPIX * 4);
    float*    bufA = (float*)alloc((size_t)BB * 4 * CC * NPIX * 4);
    float*    bufB = (float*)alloc((size_t)BB * 2 * CC * NPIX * 4);
    _Float16* actT = (_Float16*)alloc((size_t)BB * NPIX * 4 * CC * 2);  // (B,N,Cin<=4C) f16
    _Float16* xnT  = (_Float16*)alloc((size_t)BB * NPIX * CC * 2);      // (B,N,C) f16
    float*    sq   = (float*)alloc((size_t)BB * NPIX * 4);
    int*      idx  = (int*)alloc((size_t)BB * NPIX * KNN * 4);

    auto cvt = [&](const float* src, _Float16* dst, int n) {
        cvt_f16_kernel<<<(n + 255) / 256, 256, 0, stream>>>(src, dst, n);
    };
    cvt(g1_fc1_w, w16_g1_fc1, CC * CC);
    cvt(g1_mr_w,  w16_g1_mr,  2 * CC * 2 * CC);
    cvt(g1_fc2_w, w16_g1_fc2, CC * 2 * CC);
    cvt(g2_fc1_w, w16_g2_fc1, CC * CC);
    cvt(g2_mr_w,  w16_g2_mr,  2 * CC * 2 * CC);
    cvt(g2_fc2_w, w16_g2_fc2, CC * 2 * CC);
    cvt(f1_w1, w16_f1_w1, 4 * CC * CC);
    cvt(f1_w2, w16_f1_w2, CC * 4 * CC);
    cvt(f2_w1, w16_f2_w1, 4 * CC * CC);
    cvt(f2_w2, w16_f2_w2, CC * 4 * CC);

    // state <- x (never mutate inputs)
    hipMemcpyAsync(cur, x, (size_t)BB * CC * NPIX * 4, hipMemcpyDeviceToDevice, stream);

    launch_grapher(rp, cur, bufA, bufB, actT, xnT, sq, idx,
                   w16_g1_fc1, g1_fc1_b, w16_g1_mr, g1_mr_b, w16_g1_fc2, g1_fc2_b, stream);
    launch_ffn(cur, bufA, bufB, actT, w16_f1_w1, f1_b1, w16_f1_w2, f1_b2, stream);

    // y = relu(inorm(y))
    inorm_kernel<<<BB * CC, 256, 0, stream>>>(cur, cur, nullptr, 2);

    launch_grapher(rp, cur, bufA, bufB, actT, xnT, sq, idx,
                   w16_g2_fc1, g2_fc1_b, w16_g2_mr, g2_mr_b, w16_g2_fc2, g2_fc2_b, stream);
    launch_ffn(cur, bufA, bufB, actT, w16_f2_w1, f2_b1, w16_f2_w2, f2_b2, stream);

    // out = x + inorm(y)
    inorm_kernel<<<BB * CC, 256, 0, stream>>>(cur, (float*)d_out, x, 0);
}
